// MyBahdanauAttention_78271484003128
// MI455X (gfx1250) — matrix operand order repaired and checked by reading, not compiled, not run on hardware
//
#include <hip/hip_runtime.h>
#include <hip/hip_bf16.h>
#include <math.h>

// Problem sizes (fixed by the reference)
#define LSEQ 2048
#define BB   32
#define HH   1024
#define MM   (LSEQ * BB)   // 65536 rows of the big GEMM

typedef __attribute__((ext_vector_type(16))) __bf16 v16bf;
typedef __attribute__((ext_vector_type(8)))  float  v8f;

__device__ __forceinline__ __bf16 to_bf16(float f) { return (__bf16)f; }

// ---------------------------------------------------------------------------
// Kernel 0: w1 (f32, [o][h]) -> w1t (bf16, [h][o]).  2 MB result, L2-resident.
// ---------------------------------------------------------------------------
__global__ void convert_w1t_kernel(const float* __restrict__ w1,
                                   __bf16* __restrict__ w1t) {
    int idx = blockIdx.x * blockDim.x + threadIdx.x;   // over HH*HH
    int h = idx >> 10;          // / HH
    int o = idx & (HH - 1);     // % HH
    w1t[idx] = to_bf16(w1[(size_t)o * HH + h]);
}

// ---------------------------------------------------------------------------
// Kernel 1: proj_st[b][o] = sum_h state[b][h] * w2[o][h] + w2_b[o]
// state = last_decoder_state[0] -> first B*H floats of that tensor.
// ---------------------------------------------------------------------------
__global__ void proj_state_kernel(const float* __restrict__ state,
                                  const float* __restrict__ w2,
                                  const float* __restrict__ w2b,
                                  float* __restrict__ pst) {
    int b = blockIdx.x;
    int o = blockIdx.y * blockDim.x + threadIdx.x;
    const float* s    = state + (size_t)b * HH;
    const float* wrow = w2    + (size_t)o * HH;
    float acc = 0.f;
    for (int h = 0; h < HH; ++h) acc += s[h] * wrow[h];
    pst[(size_t)b * HH + o] = acc + w2b[o];
}

// ---------------------------------------------------------------------------
// Kernel 2 (the big one): fused  score[r] = v . tanh(enc[r,:]@w1^T + b1 + pst[b])
// Workgroup: 32 rows (aligned -> local row == batch index b), 8 waves cover
// N = 1024 (128 cols each as 8 N-tiles), 2 M-tiles per wave.
// K loop: 32 per step via v_wmma_f32_16x16x32_bf16, f32 accumulation.
// ---------------------------------------------------------------------------
__global__ __launch_bounds__(256)
void fused_proj_score_kernel(const float* __restrict__ enc,
                             const __bf16* __restrict__ w1t,   // [k][n] bf16
                             const float* __restrict__ w1b,
                             const float* __restrict__ pst,    // [b][n]
                             const float* __restrict__ vw,     // [n]
                             const float* __restrict__ vb,     // [1]
                             float* __restrict__ score) {      // [MM]
    __shared__ float red[32][128];

    const int tid = threadIdx.x;
    const int wv  = tid >> 5;          // wave 0..7
    const int ln  = tid & 31;          // lane
    const int l15 = ln & 15;
    const bool hi = (ln >= 16);
    const int r0  = blockIdx.x * 32;   // first row of this workgroup
    const int n0  = wv * 128;          // first column of this wave

    v8f acc[2][8];
#pragma unroll
    for (int mt = 0; mt < 2; ++mt)
#pragma unroll
        for (int t = 0; t < 8; ++t)
#pragma unroll
            for (int j = 0; j < 8; ++j) acc[mt][t][j] = 0.f;

    // A rows for this lane (16-bit A-matrix layout: lanes 0-15 row M=l15,
    // lanes 16-31 same row, complementary K halves)
    const float* arow[2];
    arow[0] = enc + (size_t)(r0 + l15) * HH;
    arow[1] = enc + (size_t)(r0 + 16 + l15) * HH;

    for (int k0 = 0; k0 < HH; k0 += 32) {
        // ---- B fragments: lane holds column N=l15 of each tile, K = {kb..kb+7, kb+16..kb+23}
        const __bf16* brow = w1t + (size_t)(k0 + (hi ? 8 : 0)) * HH + n0 + l15;
        v16bf bfrag[8];
#pragma unroll
        for (int t = 0; t < 8; ++t)
#pragma unroll
            for (int s = 0; s < 8; ++s) {
                bfrag[t][s]     = brow[(size_t)s * HH + t * 16];
                bfrag[t][s + 8] = brow[(size_t)(s + 16) * HH + t * 16];
            }

        // ---- A fragments: convert f32 -> bf16 in-register
        // lane<16: K = {k0..k0+7, k0+16..k0+23}; lane>=16: shifted by 8
        v16bf afrag[2];
        const int kb = k0 + (hi ? 8 : 0);
#pragma unroll
        for (int mt = 0; mt < 2; ++mt) {
            const float* ar = arow[mt];
            float4 x0 = *(const float4*)(ar + kb);
            float4 x1 = *(const float4*)(ar + kb + 4);
            float4 y0 = *(const float4*)(ar + kb + 16);
            float4 y1 = *(const float4*)(ar + kb + 20);
            float lo[8] = {x0.x, x0.y, x0.z, x0.w, x1.x, x1.y, x1.z, x1.w};
            float hi8[8] = {y0.x, y0.y, y0.z, y0.w, y1.x, y1.y, y1.z, y1.w};
            v16bf a;
#pragma unroll
            for (int i = 0; i < 8; ++i) {
                a[i]     = to_bf16(lo[i]);
                a[i + 8] = to_bf16(hi8[i]);
            }
            afrag[mt] = a;
        }

        // Prefetch next K-step (emits global_prefetch_b8)
        if (k0 + 32 < HH) {
            __builtin_prefetch(brow + (size_t)32 * HH, 0, 0);
            __builtin_prefetch(arow[0] + kb + 32, 0, 0);
            __builtin_prefetch(arow[1] + kb + 32, 0, 0);
        }

        // ---- 16 WMMAs per K-step
#pragma unroll
        for (int mt = 0; mt < 2; ++mt)
#pragma unroll
            for (int t = 0; t < 8; ++t)
                acc[mt][t] = __builtin_amdgcn_wmma_f32_16x16x32_bf16(
                    false, afrag[mt], false, bfrag[t],
                    (short)0, acc[mt][t], false, false);
    }

    // ---- Epilogue: + w1_b + proj_st, tanh, dot with v, reduce over N.
    // C layout: lanes 0-15 -> N=l15, M=vgpr j; lanes 16-31 -> N=l15, M=j+8.
    // r0 is a multiple of 32 and B==32, so batch index b == local row.
    float spart[16];
#pragma unroll
    for (int i = 0; i < 16; ++i) spart[i] = 0.f;

#pragma unroll
    for (int mt = 0; mt < 2; ++mt)
#pragma unroll
        for (int t = 0; t < 8; ++t) {
            const int n = n0 + t * 16 + l15;
            const float w1bn = w1b[n];
            const float vn = vw[n];
#pragma unroll
            for (int j = 0; j < 8; ++j) {
                const int row = mt * 16 + (hi ? j + 8 : j);     // == b
                float val = acc[mt][t][j] + w1bn + pst[(size_t)row * HH + n];
                spart[mt * 8 + j] += tanhf(val) * vn;
            }
        }

#pragma unroll
    for (int mt = 0; mt < 2; ++mt)
#pragma unroll
        for (int j = 0; j < 8; ++j) {
            const int row = mt * 16 + (hi ? j + 8 : j);
            red[row][wv * 16 + l15] = spart[mt * 8 + j];
        }
    __syncthreads();

    if (tid < 32) {
        float s = vb[0];
        for (int i = 0; i < 128; ++i) s += red[tid][i];
        score[r0 + tid] = s;   // r = l*B + b; row index matches reference flat order
    }
}

// ---------------------------------------------------------------------------
// Kernel 3: softmax over L per batch column; writes weights into d_out.
// ---------------------------------------------------------------------------
__global__ void softmax_kernel(const float* __restrict__ score,
                               float* __restrict__ outw) {   // d_out + B*H
    __shared__ float red[256];
    const int b = blockIdx.x;
    const int tid = threadIdx.x;

    float m = -INFINITY;
    for (int l = tid; l < LSEQ; l += 256)
        m = fmaxf(m, score[l * BB + b]);
    red[tid] = m;
    __syncthreads();
    for (int s = 128; s > 0; s >>= 1) {
        if (tid < s) red[tid] = fmaxf(red[tid], red[tid + s]);
        __syncthreads();
    }
    m = red[0];
    __syncthreads();

    float sum = 0.f;
    for (int l = tid; l < LSEQ; l += 256) {
        float e = __expf(score[l * BB + b] - m);
        outw[l * BB + b] = e;
        sum += e;
    }
    red[tid] = sum;
    __syncthreads();
    for (int s = 128; s > 0; s >>= 1) {
        if (tid < s) red[tid] += red[tid + s];
        __syncthreads();
    }
    const float inv = 1.0f / red[0];

    for (int l = tid; l < LSEQ; l += 256)
        outw[l * BB + b] *= inv;
}

// ---------------------------------------------------------------------------
// Kernel 4: context partials over L-chunks (deterministic, no float atomics)
// part[lc][b][h] = sum_{l in chunk lc} w[l,b] * enc[l,b,h]
// ---------------------------------------------------------------------------
__global__ void ctx_partial_kernel(const float* __restrict__ enc,
                                   const float* __restrict__ wgt,  // [L*B]
                                   float* __restrict__ part) {
    const int b  = blockIdx.x;
    const int hc = blockIdx.y;
    const int lc = blockIdx.z;
    const int h  = hc * 256 + threadIdx.x;
    float acc = 0.f;
    const int lbeg = lc * (LSEQ / 8), lend = lbeg + (LSEQ / 8);
    for (int l = lbeg; l < lend; ++l) {
        const int r = l * BB + b;
        acc += wgt[r] * enc[(size_t)r * HH + h];
    }
    part[((size_t)lc * BB + b) * HH + h] = acc;
}

__global__ void ctx_reduce_kernel(const float* __restrict__ part,
                                  float* __restrict__ ctx) {   // d_out
    const int b = blockIdx.x;
    const int h = blockIdx.y * 256 + threadIdx.x;
    float s = 0.f;
    for (int lc = 0; lc < 8; ++lc)
        s += part[((size_t)lc * BB + b) * HH + h];
    ctx[(size_t)b * HH + h] = s;
}

// ---------------------------------------------------------------------------
extern "C" void kernel_launch(void* const* d_in, const int* in_sizes, int n_in,
                              void* d_out, int out_size, void* d_ws, size_t ws_size,
                              hipStream_t stream) {
    const float* enc   = (const float*)d_in[0];  // [L,B,H]
    const float* state = (const float*)d_in[1];  // [2,1,B,H] -> use first B*H
    const float* w1_w  = (const float*)d_in[2];  // [H,H]
    const float* w1_b  = (const float*)d_in[3];  // [H]
    const float* w2_w  = (const float*)d_in[4];  // [H,H]
    const float* w2_b  = (const float*)d_in[5];  // [H]
    const float* v_w   = (const float*)d_in[6];  // [1,H]
    const float* v_b   = (const float*)d_in[7];  // [1]
    (void)in_sizes; (void)n_in; (void)out_size; (void)ws_size;

    float* ctx_out = (float*)d_out;                       // [B,H]
    float* wgt_out = (float*)d_out + (size_t)BB * HH;     // [L,B]

    // Workspace layout (bytes, 256B-aligned regions)
    char* ws = (char*)d_ws;
    __bf16* w1t  = (__bf16*)(ws);                               // 2 MB
    float*  pst  = (float*)(ws + (size_t)HH * HH * 2);          // 128 KB
    float*  scr  = (float*)(ws + (size_t)HH * HH * 2 + 131072); // 256 KB
    float*  part = (float*)(ws + (size_t)HH * HH * 2 + 131072 + 262144); // 1 MB

    // 0) w1 -> bf16 transposed [k][n]
    convert_w1t_kernel<<<(HH * HH) / 256, 256, 0, stream>>>(w1_w, w1t);

    // 1) proj_st
    proj_state_kernel<<<dim3(BB, HH / 256), 256, 0, stream>>>(state, w2_w, w2_b, pst);

    // 2) fused GEMM + bias + tanh + v-dot  (2048 workgroups x 8 waves)
    fused_proj_score_kernel<<<MM / 32, 256, 0, stream>>>(
        enc, w1t, w1_b, pst, v_w, v_b, scr);

    // 3) softmax over L -> weights in d_out
    softmax_kernel<<<BB, 256, 0, stream>>>(scr, wgt_out);

    // 4) context = sum_l w * enc  (deterministic two-phase reduction)
    ctx_partial_kernel<<<dim3(BB, HH / 256, 8), 256, 0, stream>>>(enc, wgt_out, part);
    ctx_reduce_kernel<<<dim3(BB, HH / 256), 256, 0, stream>>>(part, ctx_out);
}
